// SingleHeadAttention_62156766707961
// MI455X (gfx1250) — compile-verified, matrix operand order checked by reading, not run on hardware
//
#include <hip/hip_runtime.h>

typedef __attribute__((ext_vector_type(16))) __bf16 v16bf;
typedef __attribute__((ext_vector_type(8)))  float  v8f;

#define B_   8
#define S_   2048
#define H_   768
#define D_   64
#define ROWS (B_ * S_)

// k-index held by element j of a 16-bit A/B fragment for lane-half h
// (ISA 7.12.2: VGPR0..3 hold K=h*8+0..7, VGPR4..7 hold K=16+h*8+0..7)
__device__ __forceinline__ int kmap(int j, int h) {
  return (j < 8) ? (h * 8 + j) : (16 + h * 8 + (j - 8));
}

// hardware round-to-nearest-even f32->bf16 (v_cvt_*bf16_f32)
__device__ __forceinline__ __bf16 f2bf(float f) { return static_cast<__bf16>(f); }

// reductions across the 16 lanes holding one matrix row (xor masks stay in-half)
__device__ __forceinline__ float rmax16(float v) {
  v = fmaxf(v, __shfl_xor(v, 1, 32));
  v = fmaxf(v, __shfl_xor(v, 2, 32));
  v = fmaxf(v, __shfl_xor(v, 4, 32));
  v = fmaxf(v, __shfl_xor(v, 8, 32));
  return v;
}
__device__ __forceinline__ float rsum16(float v) {
  v += __shfl_xor(v, 1, 32);
  v += __shfl_xor(v, 2, 32);
  v += __shfl_xor(v, 4, 32);
  v += __shfl_xor(v, 8, 32);
  return v;
}

#define WMMA_BF16(A, Bm, C) \
  __builtin_amdgcn_wmma_f32_16x16x32_bf16(false, (A), false, (Bm), (short)0, (C), false, false)

// ---------------------------------------------------------------------------
// Kernel 1: fused QKV projection.  [16384 x 768] * [768 x 64] (x3) in bf16
// WMMA with f32 accumulate.  One pass over x; q/k/v written bf16 to workspace.
// Block = 128 threads (4 waves), M-tile = 64 rows, K streamed in 32-chunks.
// W staged TRANSPOSED ([n][k]) so B-fragments are contiguous 16B runs.
// ---------------------------------------------------------------------------
__global__ __launch_bounds__(128) void qkv_proj_kernel(
    const float* __restrict__ x,
    const float* __restrict__ Wq, const float* __restrict__ bq,
    const float* __restrict__ Wk, const float* __restrict__ bk,
    const float* __restrict__ Wv, const float* __restrict__ bv,
    __bf16* __restrict__ qo, __bf16* __restrict__ ko, __bf16* __restrict__ vo)
{
  __shared__ __attribute__((aligned(16))) __bf16 lds_x[64 * 32];        // 4 KB  [m][k]
  __shared__ __attribute__((aligned(16))) __bf16 lds_wt[3][64 * 32];    // 12 KB [n][k]

  const int tid  = threadIdx.x;
  const int lane = tid & 31;
  const int w    = tid >> 5;        // wave 0..3 -> 16-row sub-tile
  const int nl   = lane & 15;
  const int h    = lane >> 4;
  const int row0 = blockIdx.x * 64;

  const float* Ws[3] = {Wq, Wk, Wv};

  v8f acc[3][4];
  #pragma unroll
  for (int p = 0; p < 3; ++p)
    #pragma unroll
    for (int nt = 0; nt < 4; ++nt)
      #pragma unroll
      for (int r = 0; r < 8; ++r) acc[p][nt][r] = 0.f;

  for (int k0 = 0; k0 < H_; k0 += 32) {
    // stage x tile [64 x 32]: float4 global reads, contiguous bf16 LDS stores
    for (int idx = tid; idx < 512; idx += 128) {
      int r = idx >> 3, c4 = (idx & 7) * 4;
      float4 xv = *reinterpret_cast<const float4*>(&x[(size_t)(row0 + r) * H_ + k0 + c4]);
      lds_x[r * 32 + c4 + 0] = f2bf(xv.x);
      lds_x[r * 32 + c4 + 1] = f2bf(xv.y);
      lds_x[r * 32 + c4 + 2] = f2bf(xv.z);
      lds_x[r * 32 + c4 + 3] = f2bf(xv.w);
    }
    // stage Wq/Wk/Wv chunks transposed: [64 n][32 k]
    #pragma unroll
    for (int p = 0; p < 3; ++p)
      for (int idx = tid; idx < 512; idx += 128) {
        int r = idx >> 4, c4 = (idx & 15) * 4;    // r = k-row, c4 = n-col group
        float4 wv = *reinterpret_cast<const float4*>(&Ws[p][(size_t)(k0 + r) * D_ + c4]);
        lds_wt[p][(c4 + 0) * 32 + r] = f2bf(wv.x);
        lds_wt[p][(c4 + 1) * 32 + r] = f2bf(wv.y);
        lds_wt[p][(c4 + 2) * 32 + r] = f2bf(wv.z);
        lds_wt[p][(c4 + 3) * 32 + r] = f2bf(wv.w);
      }
    __syncthreads();

    // A fragment: 16 rows of x, K=32 (two contiguous 8-element runs per lane)
    v16bf a;
    #pragma unroll
    for (int j = 0; j < 16; ++j)
      a[j] = lds_x[(w * 16 + nl) * 32 + kmap(j, h)];

    #pragma unroll
    for (int p = 0; p < 3; ++p) {
      #pragma unroll
      for (int nt = 0; nt < 4; ++nt) {
        v16bf bm;
        #pragma unroll
        for (int j = 0; j < 16; ++j)
          bm[j] = lds_wt[p][(nt * 16 + nl) * 32 + kmap(j, h)];   // contiguous runs
        acc[p][nt] = WMMA_BF16(a, bm, acc[p][nt]);
      }
    }
    __syncthreads();
  }

  const float* biases[3] = {bq, bk, bv};
  __bf16* outs[3] = {qo, ko, vo};
  #pragma unroll
  for (int p = 0; p < 3; ++p) {
    #pragma unroll
    for (int nt = 0; nt < 4; ++nt) {
      float bb = biases[p][nt * 16 + nl];
      #pragma unroll
      for (int r = 0; r < 8; ++r) {
        int row = row0 + w * 16 + r + 8 * h;    // C-layout row = r + 8*half
        outs[p][(size_t)row * D_ + nt * 16 + nl] = f2bf(acc[p][nt][r] + bb);
      }
    }
  }
}

// ---------------------------------------------------------------------------
// Kernel 2: causal flash attention.  Block = 256 threads (8 waves) owns a
// 128-row query tile; streams 128-key K/V blocks through LDS; online softmax
// in f32; S = Q*K^T and O += P*V both via v_wmma_f32_16x16x32_bf16.
// K block is staged with GLOBAL_LOAD_ASYNC_TO_LDS_B128 (ASYNCcnt path,
// no VGPR round-trip); V needs a transpose so it goes through VGPRs.
// NOTE: reference applies softmax FIRST, then /sqrt(64) -> fold 1/8 into
// the final normalization.
// ---------------------------------------------------------------------------
__global__ __launch_bounds__(256) void flash_attn_kernel(
    const __bf16* __restrict__ q, const __bf16* __restrict__ k,
    const __bf16* __restrict__ v, float* __restrict__ out)
{
  __shared__ __attribute__((aligned(16))) __bf16 lds_k [128 * 64];      // 16 KB [key][d]
  __shared__ __attribute__((aligned(16))) __bf16 lds_vt[64 * 128];      // 16 KB [d][key]
  __shared__ __attribute__((aligned(16))) __bf16 lds_p [8 * 16 * 128];  // 32 KB per-wave P

  const int tid  = threadIdx.x;
  const int lane = tid & 31;
  const int w    = tid >> 5;        // wave 0..7 -> 16 query rows
  const int nl   = lane & 15;
  const int h    = lane >> 4;
  const int q0   = blockIdx.x * 128;
  const int b    = blockIdx.y;

  const __bf16* qb = q + (size_t)b * S_ * D_;
  const __bf16* kg = k + (size_t)b * S_ * D_;
  const __bf16* vg = v + (size_t)b * S_ * D_;

  // Q fragments for this wave's 16 rows (K = 64 -> two 16x32 fragments),
  // contiguous 16-byte runs -> global_load_b128
  v16bf aq0, aq1;
  #pragma unroll
  for (int j = 0; j < 16; ++j) {
    int kk = kmap(j, h);
    aq0[j] = qb[(size_t)(q0 + w * 16 + nl) * D_ + kk];
    aq1[j] = qb[(size_t)(q0 + w * 16 + nl) * D_ + 32 + kk];
  }

  v8f acc[4];
  float m_r[8], l_r[8];
  #pragma unroll
  for (int nt = 0; nt < 4; ++nt)
    #pragma unroll
    for (int r = 0; r < 8; ++r) acc[nt][r] = 0.f;
  #pragma unroll
  for (int r = 0; r < 8; ++r) { m_r[r] = -3.0e38f; l_r[r] = 0.f; }

  const int nkb = q0 / 128 + 1;                  // causal: only blocks <= diagonal
  for (int kbi = 0; kbi < nkb; ++kbi) {
    // --- K block: async DMA global -> LDS, 16B per lane per issue ---------
    #pragma unroll
    for (int it = 0; it < 4; ++it) {
      int idx = tid + it * 256;                  // b128 chunk id (0..1023)
      unsigned loff = (unsigned)(size_t)(&lds_k[idx * 8]);
      const __bf16* gsrc = kg + (size_t)kbi * 8192 + idx * 8;
      asm volatile("global_load_async_to_lds_b128 %0, %1, off"
                   :: "v"(loff), "v"(gsrc) : "memory");
    }
    // --- V block: uint4 global reads, scattered b16 stores (transpose) ----
    for (int idx = tid; idx < 1024; idx += 256) {
      uint4 vv = *reinterpret_cast<const uint4*>(&vg[(size_t)kbi * 8192 + idx * 8]);
      __bf16 tmp[8];
      __builtin_memcpy(tmp, &vv, 16);
      int key = idx >> 3, d0 = (idx & 7) * 8;
      #pragma unroll
      for (int e = 0; e < 8; ++e) lds_vt[(d0 + e) * 128 + key] = tmp[e];
    }
    if (kbi + 1 < nkb)                            // global_prefetch_b8 next K block
      __builtin_prefetch(kg + (size_t)(kbi + 1) * 8192 + tid * 32, 0, 1);
    asm volatile("s_wait_asynccnt 0" ::: "memory");
    __syncthreads();

    // scores S[16 x 128]: 8 N-tiles x 2 K-chunks of WMMA
    v8f s[8];
    #pragma unroll
    for (int t = 0; t < 8; ++t) {
      v16bf b0, b1;
      #pragma unroll
      for (int j = 0; j < 16; ++j) {
        int kk = kmap(j, h);
        b0[j] = lds_k[(t * 16 + nl) * 64 + kk];        // K^T: contiguous runs
        b1[j] = lds_k[(t * 16 + nl) * 64 + 32 + kk];
      }
      v8f st;
      #pragma unroll
      for (int r = 0; r < 8; ++r) st[r] = 0.f;
      st = WMMA_BF16(aq0, b0, st);
      st = WMMA_BF16(aq1, b1, st);
      s[t] = st;
    }

    if (kbi == nkb - 1) {                          // diagonal block: causal mask
      #pragma unroll
      for (int t = 0; t < 8; ++t)
        #pragma unroll
        for (int r = 0; r < 8; ++r) {
          int qrow = q0 + w * 16 + r + 8 * h;
          int kcol = kbi * 128 + t * 16 + nl;
          s[t][r] = (kcol > qrow) ? -3.0e38f : s[t][r];
        }
    }

    // online softmax update (rows live in 16-lane halves)
    #pragma unroll
    for (int r = 0; r < 8; ++r) {
      float mloc = s[0][r];
      #pragma unroll
      for (int t = 1; t < 8; ++t) mloc = fmaxf(mloc, s[t][r]);
      mloc = rmax16(mloc);
      float m_new = fmaxf(m_r[r], mloc);
      float alpha = __expf(m_r[r] - m_new);
      float psum = 0.f;
      #pragma unroll
      for (int t = 0; t < 8; ++t) {
        float p = __expf(s[t][r] - m_new);
        s[t][r] = p;
        psum += p;
      }
      psum   = rsum16(psum);
      l_r[r] = l_r[r] * alpha + psum;
      m_r[r] = m_new;
      #pragma unroll
      for (int nt = 0; nt < 4; ++nt) acc[nt][r] *= alpha;
      #pragma unroll
      for (int t = 0; t < 8; ++t)                  // C-layout -> A-layout via LDS
        lds_p[w * 2048 + (r + 8 * h) * 128 + t * 16 + nl] = f2bf(s[t][r]);
    }

    // O += P[16x128] * V[128x64]: 4 K-chunks x 4 N-tiles of WMMA
    #pragma unroll
    for (int kc = 0; kc < 4; ++kc) {
      v16bf ap;
      #pragma unroll
      for (int j = 0; j < 16; ++j)
        ap[j] = lds_p[w * 2048 + nl * 128 + kc * 32 + kmap(j, h)];
      #pragma unroll
      for (int nt = 0; nt < 4; ++nt) {
        v16bf bv_;
        #pragma unroll
        for (int j = 0; j < 16; ++j)
          bv_[j] = lds_vt[(nt * 16 + nl) * 128 + kc * 32 + kmap(j, h)];
        acc[nt] = WMMA_BF16(ap, bv_, acc[nt]);
      }
    }
    __syncthreads();
  }

  // epilogue: softmax normalize (reciprocal once per row), then /sqrt(64)
  float inv_l[8];
  #pragma unroll
  for (int r = 0; r < 8; ++r) inv_l[r] = 1.0f / (l_r[r] * 8.0f);
  #pragma unroll
  for (int nt = 0; nt < 4; ++nt)
    #pragma unroll
    for (int r = 0; r < 8; ++r) {
      int row = q0 + w * 16 + r + 8 * h;
      out[((size_t)b * S_ + row) * D_ + nt * 16 + nl] = acc[nt][r] * inv_l[r];
    }
}

extern "C" void kernel_launch(void* const* d_in, const int* in_sizes, int n_in,
                              void* d_out, int out_size, void* d_ws, size_t ws_size,
                              hipStream_t stream) {
  (void)in_sizes; (void)n_in; (void)out_size; (void)ws_size;
  const float* x  = (const float*)d_in[0];
  const float* Wq = (const float*)d_in[1];
  const float* bq = (const float*)d_in[2];
  const float* Wk = (const float*)d_in[3];
  const float* bk = (const float*)d_in[4];
  const float* Wv = (const float*)d_in[5];
  const float* bv = (const float*)d_in[6];
  float* out = (float*)d_out;

  const size_t n = (size_t)ROWS * D_;            // 1 Mi elements per tensor
  __bf16* qw = (__bf16*)d_ws;                    // 2 MB
  __bf16* kw = qw + n;                           // 2 MB
  __bf16* vw = kw + n;                           // 2 MB (ws >= 6 MB)

  qkv_proj_kernel<<<dim3(ROWS / 64), 128, 0, stream>>>(
      x, Wq, bq, Wk, bk, Wv, bv, qw, kw, vw);
  flash_attn_kernel<<<dim3(S_ / 128, B_), 256, 0, stream>>>(qw, kw, vw, out);
}
